// CausalAlibiAttn_28003186770033
// MI455X (gfx1250) — matrix operand order repaired and checked by reading, not compiled, not run on hardware
//
#include <hip/hip_runtime.h>
#include <math.h>

// ---------------------------------------------------------------------------
// CDNA5 (gfx1250) causal-ALiBi GQA attention, bf16 WMMA pipeline. wave32.
//   pass 1: pack weights f32 -> bf16 in WMMA-B-operand-ready tiles (once)
//   pass 2: Q/K projections (Q,K row-major bf16), V projection stored transposed
//   pass 3: flash attention (WMMA QK^T, online softmax, WMMA P.V)
//   pass 4: output projection -> f32
// ---------------------------------------------------------------------------

typedef __attribute__((ext_vector_type(16))) __bf16 v16bf;
typedef __attribute__((ext_vector_type(8)))  __bf16 v8bf;
typedef __attribute__((ext_vector_type(8)))  float  v8f;

#define SEQ   4096
#define DM    1024
#define NH    16
#define NKV   4
#define DH    64
#define DKV   (NKV * DH)   // 256

static __device__ __forceinline__ v8f wmma_bf16(v16bf a, v16bf b, v8f c) {
  return __builtin_amdgcn_wmma_f32_16x16x32_bf16(false, a, false, b, (short)0, c,
                                                 false, false);
}

// ---- A operand 16x32 bf16 (this file's contiguous k order, not AMD's table): lane<16 row M=lane holds
// K {0..15}; lane>=16 row M=lane-16 holds K {16..31}.
static __device__ __forceinline__
v16bf load_a_f32(const float* A, int ld, int row0, int k0, int lrow, int lhalf) {
  const float* p = A + (size_t)(row0 + lrow) * ld + k0 + 16 * lhalf;
  v8f x0 = *(const v8f*)(p);        // 32B coalesced
  v8f x1 = *(const v8f*)(p + 8);    // 32B coalesced
  v16bf a;
#pragma unroll
  for (int e = 0; e < 8; ++e) { a[e] = (__bf16)x0[e]; a[8 + e] = (__bf16)x1[e]; }
  return a;
}

static __device__ __forceinline__
v16bf load_a_bf16(const __bf16* A, int ld, int row0, int k0, int lrow, int lhalf) {
  const __bf16* p = A + (size_t)(row0 + lrow) * ld + k0 + 16 * lhalf;
  v8bf lo = *(const v8bf*)(p);
  v8bf hi = *(const v8bf*)(p + 8);
  return __builtin_shufflevector(lo, hi, 0, 1, 2, 3, 4, 5, 6, 7,
                                 8, 9, 10, 11, 12, 13, 14, 15);
}

// ---- B operand 32x16 from an [n][k] row-major bf16 matrix (i.e. the memory
// matrix holds B transposed): lane has N=n0+lrow, elements e -> K=k0+16*lhalf+e,
// which is one contiguous 32B load per lane.
static __device__ __forceinline__
v16bf load_bT(const __bf16* M, int ld, int n0, int k0, int lrow, int lhalf) {
  return *(const v16bf*)(M + (size_t)(n0 + lrow) * ld + k0 + 16 * lhalf);
}

// ---------------------------------------------------------------------------
// Weight pack: W_f32[K][N] -> Wp tiles [kt][nt][lane][16] bf16 so that a GEMM
// B-operand is exactly one contiguous v16bf per lane. One wave per 32x16 tile.
// ---------------------------------------------------------------------------
__global__ __launch_bounds__(32)
void pack_w(const float* __restrict__ W, __bf16* __restrict__ Wp, int N) {
  const int lane  = threadIdx.x & 31;
  const int lrow  = lane & 15;
  const int lhalf = lane >> 4;
  const int kt = blockIdx.x, nt = blockIdx.y;
  const int ntiles = N >> 4;
  const float* p = W + (size_t)(kt * 32 + 16 * lhalf) * N + nt * 16 + lrow;
  v16bf b;
#pragma unroll
  for (int e = 0; e < 16; ++e) b[e] = (__bf16)p[(size_t)e * N];
  *((v16bf*)Wp + ((size_t)(kt * ntiles + nt) * 32 + lane)) = b;
}

// ---------------------------------------------------------------------------
// GEMM: C = bf16(A_f32[M,K] @ W + bias), W pre-packed bf16. Wave = 32x64 tile.
// TRANS=false: C[M][N] row-major; TRANS=true: C[N][M] (ld = SEQ) for V^T.
// grid = (M/32, N/64), block = 32.
// ---------------------------------------------------------------------------
template <bool TRANS>
__global__ __launch_bounds__(32)
void gemm_f32p_bf16(const float* __restrict__ A, const __bf16* __restrict__ Wp,
                    const float* __restrict__ bias, __bf16* __restrict__ C,
                    int N, int K) {
  const int lane  = threadIdx.x & 31;
  const int lrow  = lane & 15;
  const int lhalf = lane >> 4;
  const int row0  = blockIdx.x * 32;
  const int n0    = blockIdx.y * 64;
  const int ntiles = N >> 4;

  v8f acc[2][4] = {};
  for (int kb = 0; kb < K; kb += 32) {
    v16bf a0 = load_a_f32(A, K, row0,      kb, lrow, lhalf);
    v16bf a1 = load_a_f32(A, K, row0 + 16, kb, lrow, lhalf);
    const v16bf* wp = (const v16bf*)Wp +
                      ((size_t)(kb >> 5) * ntiles + (n0 >> 4)) * 32 + lane;
#pragma unroll
    for (int t = 0; t < 4; ++t) {
      v16bf b = wp[t * 32];
      acc[0][t] = wmma_bf16(a0, b, acc[0][t]);
      acc[1][t] = wmma_bf16(a1, b, acc[1][t]);
    }
  }
#pragma unroll
  for (int mi = 0; mi < 2; ++mi) {
#pragma unroll
    for (int t = 0; t < 4; ++t) {
      const int col = n0 + t * 16 + lrow;
      const float bv = bias[col];
#pragma unroll
      for (int r = 0; r < 8; ++r) {
        const int row = row0 + mi * 16 + r + 8 * lhalf;
        const __bf16 val = (__bf16)(acc[mi][t][r] + bv);
        if (TRANS) C[(size_t)col * SEQ + row] = val;
        else       C[(size_t)row * N + col]   = val;
      }
    }
  }
}

// ---------------------------------------------------------------------------
// Output projection: C_f32 = A_bf16[M,K] @ Wp + bias. Wave = 32x64 tile.
// ---------------------------------------------------------------------------
__global__ __launch_bounds__(32)
void gemm_bf16p_f32(const __bf16* __restrict__ A, const __bf16* __restrict__ Wp,
                    const float* __restrict__ bias, float* __restrict__ C,
                    int N, int K) {
  const int lane  = threadIdx.x & 31;
  const int lrow  = lane & 15;
  const int lhalf = lane >> 4;
  const int row0  = blockIdx.x * 32;
  const int n0    = blockIdx.y * 64;
  const int ntiles = N >> 4;

  v8f acc[2][4] = {};
  for (int kb = 0; kb < K; kb += 32) {
    v16bf a0 = load_a_bf16(A, K, row0,      kb, lrow, lhalf);
    v16bf a1 = load_a_bf16(A, K, row0 + 16, kb, lrow, lhalf);
    const v16bf* wp = (const v16bf*)Wp +
                      ((size_t)(kb >> 5) * ntiles + (n0 >> 4)) * 32 + lane;
#pragma unroll
    for (int t = 0; t < 4; ++t) {
      v16bf b = wp[t * 32];
      acc[0][t] = wmma_bf16(a0, b, acc[0][t]);
      acc[1][t] = wmma_bf16(a1, b, acc[1][t]);
    }
  }
#pragma unroll
  for (int mi = 0; mi < 2; ++mi) {
#pragma unroll
    for (int t = 0; t < 4; ++t) {
      const int col = n0 + t * 16 + lrow;
      const float bv = bias[col];
#pragma unroll
      for (int r = 0; r < 8; ++r)
        C[(size_t)(row0 + mi * 16 + r + 8 * lhalf) * N + col] = acc[mi][t][r] + bv;
    }
  }
}

// ---------------------------------------------------------------------------
// Flash attention, causal + ALiBi, GQA. One wave per (head, 16-query tile).
// K row-major [SEQ][DKV]; V transposed [DKV][SEQ] -> all WMMA operand loads
// are contiguous 32B/lane. grid = (SEQ/16, NH), block = 32.
// ---------------------------------------------------------------------------
__global__ __launch_bounds__(32)
void attn_fa(const __bf16* __restrict__ Qb, const __bf16* __restrict__ Kb,
             const __bf16* __restrict__ Vt, const float* __restrict__ slopes,
             __bf16* __restrict__ Ob) {
  __shared__ __align__(32) __bf16 sP[16 * 32];  // P: D-layout -> row-major -> A-layout

  const int lane  = threadIdx.x & 31;
  const int lrow  = lane & 15;
  const int lhalf = lane >> 4;
  const int qt    = blockIdx.x;
  const int h     = blockIdx.y;
  const int kh    = h >> 2;          // GQA: 4 q-heads per kv head
  const int q0    = qt * 16;
  const float slope = slopes[h];
  const float scale = 0.125f;        // 1/sqrt(64)

  const __bf16* Kh  = Kb + kh * DH;                 // [SEQ][DKV] column block
  const __bf16* VtH = Vt + (size_t)kh * DH * SEQ;   // rows kh*64.. of [DKV][SEQ]

  const v16bf aq0 = load_a_bf16(Qb, DM, q0, h * DH + 0,  lrow, lhalf);
  const v16bf aq1 = load_a_bf16(Qb, DM, q0, h * DH + 32, lrow, lhalf);

  v8f acc[4] = {};
  float m[8], l[8];
#pragma unroll
  for (int r = 0; r < 8; ++r) { m[r] = -INFINITY; l[r] = 0.0f; }

  const int kend = q0 + 16;
  for (int k0 = 0; k0 < kend; k0 += 32) {
    if (k0 + 32 < kend) {            // prefetch next K/V tiles (uniform branch)
      __builtin_prefetch(Kh + (size_t)(k0 + 32 + lane) * DKV, 0, 1);
      __builtin_prefetch(VtH + (size_t)(lane << 1) * SEQ + k0 + 32, 0, 1);
    }
    // ---- S = Q K^T : two 16x16 D-tiles covering keys k0..k0+31
    v8f s0 = {}, s1 = {};
    s0 = wmma_bf16(aq0, load_bT(Kh, DKV, k0,      0,  lrow, lhalf), s0);
    s0 = wmma_bf16(aq1, load_bT(Kh, DKV, k0,      32, lrow, lhalf), s0);
    s1 = wmma_bf16(aq0, load_bT(Kh, DKV, k0 + 16, 0,  lrow, lhalf), s1);
    s1 = wmma_bf16(aq1, load_bT(Kh, DKV, k0 + 16, 32, lrow, lhalf), s1);

    // ---- mask + ALiBi + online softmax (16-lane row groups)
    const int key0 = k0 + lrow;
    const int key1 = k0 + 16 + lrow;
#pragma unroll
    for (int r = 0; r < 8; ++r) {
      const int row = q0 + r + 8 * lhalf;
      float v0 = (key0 <= row) ? (s0[r] * scale + slope * (float)(key0 - row))
                               : -INFINITY;
      float v1 = (key1 <= row) ? (s1[r] * scale + slope * (float)(key1 - row))
                               : -INFINITY;
      float rmax = fmaxf(v0, v1);
#pragma unroll
      for (int off = 1; off < 16; off <<= 1)
        rmax = fmaxf(rmax, __shfl_xor(rmax, off, 32));

      const float nm   = fmaxf(m[r], rmax);
      const float corr = __expf(m[r] - nm);
      m[r] = nm;
      const float p0 = __expf(v0 - nm);
      const float p1 = __expf(v1 - nm);
      float rsum = p0 + p1;
#pragma unroll
      for (int off = 1; off < 16; off <<= 1)
        rsum += __shfl_xor(rsum, off, 32);
      l[r] = l[r] * corr + rsum;
#pragma unroll
      for (int t = 0; t < 4; ++t) acc[t][r] *= corr;

      sP[(r + 8 * lhalf) * 32 + lrow]      = (__bf16)p0;
      sP[(r + 8 * lhalf) * 32 + 16 + lrow] = (__bf16)p1;
    }
    __syncthreads();
    const v16bf ap = load_a_bf16(sP, 32, 0, 0, lrow, lhalf);
    __syncthreads();
#pragma unroll
    for (int t = 0; t < 4; ++t)
      acc[t] = wmma_bf16(ap, load_bT(VtH, SEQ, t * 16, k0, lrow, lhalf), acc[t]);
  }

#pragma unroll
  for (int t = 0; t < 4; ++t) {
#pragma unroll
    for (int r = 0; r < 8; ++r) {
      const float o = acc[t][r] / l[r];
      Ob[(size_t)(q0 + r + 8 * lhalf) * DM + h * DH + t * 16 + lrow] = (__bf16)o;
    }
  }
}

// ---------------------------------------------------------------------------
extern "C" void kernel_launch(void* const* d_in, const int* in_sizes, int n_in,
                              void* d_out, int out_size, void* d_ws, size_t ws_size,
                              hipStream_t stream) {
  (void)in_sizes; (void)n_in; (void)out_size; (void)ws_size;

  const float* qkv    = (const float*)d_in[0];
  const float* Wq     = (const float*)d_in[1];
  const float* bq     = (const float*)d_in[2];
  const float* Wk     = (const float*)d_in[3];
  const float* bk     = (const float*)d_in[4];
  const float* Wv     = (const float*)d_in[5];
  const float* bv     = (const float*)d_in[6];
  const float* Wo     = (const float*)d_in[7];
  const float* bo     = (const float*)d_in[8];
  const float* slopes = (const float*)d_in[9];
  float* out = (float*)d_out;

  char* ws = (char*)d_ws;
  size_t off = 0;
  __bf16* Qb  = (__bf16*)(ws + off); off += (size_t)SEQ * DM  * 2;  //  8 MB
  __bf16* Kb  = (__bf16*)(ws + off); off += (size_t)SEQ * DKV * 2;  //  2 MB
  __bf16* Vt  = (__bf16*)(ws + off); off += (size_t)DKV * SEQ * 2;  //  2 MB
  __bf16* Ob  = (__bf16*)(ws + off); off += (size_t)SEQ * DM  * 2;  //  8 MB
  __bf16* Wqp = (__bf16*)(ws + off); off += (size_t)DM  * DM  * 2;  //  2 MB
  __bf16* Wkp = (__bf16*)(ws + off); off += (size_t)DM  * DKV * 2;  // .5 MB
  __bf16* Wvp = (__bf16*)(ws + off); off += (size_t)DM  * DKV * 2;  // .5 MB
  __bf16* Wop = (__bf16*)(ws + off); off += (size_t)DM  * DM  * 2;  //  2 MB

  const dim3 blk(32, 1, 1);

  pack_w<<<dim3(DM / 32, DM  / 16), blk, 0, stream>>>(Wq, Wqp, DM);
  pack_w<<<dim3(DM / 32, DKV / 16), blk, 0, stream>>>(Wk, Wkp, DKV);
  pack_w<<<dim3(DM / 32, DKV / 16), blk, 0, stream>>>(Wv, Wvp, DKV);
  pack_w<<<dim3(DM / 32, DM  / 16), blk, 0, stream>>>(Wo, Wop, DM);

  gemm_f32p_bf16<false><<<dim3(SEQ / 32, DM  / 64), blk, 0, stream>>>(qkv, Wqp, bq, Qb, DM,  DM);
  gemm_f32p_bf16<false><<<dim3(SEQ / 32, DKV / 64), blk, 0, stream>>>(qkv, Wkp, bk, Kb, DKV, DM);
  gemm_f32p_bf16<true ><<<dim3(SEQ / 32, DKV / 64), blk, 0, stream>>>(qkv, Wvp, bv, Vt, DKV, DM);

  attn_fa<<<dim3(SEQ / 16, NH), blk, 0, stream>>>(Qb, Kb, Vt, slopes, Ob);

  gemm_bf16p_f32<<<dim3(SEQ / 32, DM / 64), blk, 0, stream>>>(Ob, Wop, bo, out, DM, DM);
}
